// MambaMoEModel_48730698940565
// MI455X (gfx1250) — matrix operand order repaired and checked by reading, not compiled, not run on hardware
//
#include <hip/hip_runtime.h>

typedef _Float16 half_t;
typedef __attribute__((ext_vector_type(16))) _Float16 v16h;
typedef __attribute__((ext_vector_type(8)))  _Float16 v8h;
typedef __attribute__((ext_vector_type(8)))  float    v8f;

#define B_   2
#define S_   1024
#define D_   512
#define NL_  2
#define DI_  1024
#define DS_  16
#define KC_  4
#define DFF_ 2048
#define E_   8
#define V_   32000
#define M_   (B_*S_)               // 2048 token rows
#define AUX_IDX ((long)B_*S_*V_)   // 65,536,000

#define SHUF16(lo, hi) __builtin_shufflevector(lo, hi, 0,1,2,3,4,5,6,7,8,9,10,11,12,13,14,15)
#define WMMA(a, b, c) __builtin_amdgcn_wmma_f32_16x16x32_f16(false, a, false, b, (short)0, c, false, false)

// ---------------------------------------------------------------------------
// Embedding: x[m,d] = tok_emb[ids[m],d] + pos_emb[m%S,d]
// ---------------------------------------------------------------------------
__global__ void k_embed(const int* __restrict__ ids, const float* __restrict__ tok,
                        const float* __restrict__ pos, float* __restrict__ x) {
  long i = (long)blockIdx.x * blockDim.x + threadIdx.x;   // M_*D_ threads
  int d = (int)(i & (D_ - 1));
  long m = i >> 9;
  int s = (int)(m & (S_ - 1));
  x[i] = tok[(long)ids[m] * D_ + d] + pos[(long)s * D_ + d];
}

// ---------------------------------------------------------------------------
// RMSNorm over D=512; writes fp32 (optional) and fp16 copies of the output
// ---------------------------------------------------------------------------
__global__ void k_rmsnorm(const float* __restrict__ x, const float* __restrict__ w,
                          float* __restrict__ of, half_t* __restrict__ oh) {
  __shared__ float red[256];
  int row = blockIdx.x;
  const float* xr = x + (long)row * D_;
  float s = 0.f;
  for (int d = threadIdx.x; d < D_; d += blockDim.x) { float v = xr[d]; s += v * v; }
  red[threadIdx.x] = s; __syncthreads();
  for (int st = 128; st > 0; st >>= 1) {
    if ((int)threadIdx.x < st) red[threadIdx.x] += red[threadIdx.x + st];
    __syncthreads();
  }
  float scale = rsqrtf(red[0] * (1.0f / D_) + 1e-6f);
  for (int d = threadIdx.x; d < D_; d += blockDim.x) {
    float v = xr[d] * scale * w[d];
    if (of) of[(long)row * D_ + d] = v;
    oh[(long)row * D_ + d] = (half_t)v;
  }
}

__global__ void k_cvt16(const float* __restrict__ in, half_t* __restrict__ out, long n) {
  long i = (long)blockIdx.x * blockDim.x + threadIdx.x;
  if (i < n) out[i] = (half_t)in[i];
}

__global__ void k_zero(float* __restrict__ p, int n) {
  int i = blockIdx.x * blockDim.x + threadIdx.x;
  if (i < n) p[i] = 0.f;
}

__device__ __forceinline__ float apply_act(float v, int act) {
  if (act == 1)       // gelu (tanh approx, jax default)
    v = 0.5f * v * (1.f + tanhf(0.7978845608028654f * (v + 0.044715f * v * v * v)));
  else if (act == 2)  // softplus
    v = (v > 20.f) ? v : log1pf(expf(v));
  return v;
}

// ---------------------------------------------------------------------------
// Big WMMA GEMM: C[M,N] = A[M,K] * W[N,K]^T   (A,W fp16; C fp32 or fp16)
// Register tiling: each wave owns a 32x64 output region = 2x4 WMMA tiles,
// 8 v_wmma per K-step on 2 A-fragments x 4 B-fragments.
// Requires M%32==0, N%256==0, K%32==0.
// mode 0: C(or C16) = act(val + bias)
// mode 1: C += val
// mode 2: C += gate[m*E+e]*(val+bias)
// ---------------------------------------------------------------------------
__global__ void k_gemm_big(const half_t* __restrict__ A, const half_t* __restrict__ W,
                           const float* __restrict__ bias, float* __restrict__ C,
                           half_t* __restrict__ C16,
                           const float* __restrict__ gate, int gate_e,
                           int M, int N, int K, int mode, int act) {
  int wave = threadIdx.x >> 5;
  int lane = threadIdx.x & 31;
  int n0 = (blockIdx.x * 4 + wave) * 64;
  int m0 = blockIdx.y * 32;
  int r  = lane & 15;
  int hi = lane >> 4;
  const half_t* ap0 = A + (long)(m0 + r) * K + hi * 16;       // rows m0+r, m0+16+r
  const half_t* ap1 = ap0 + (long)16 * K;
  const half_t* bp  = W + (long)(n0 + r) * K + hi * 16;       // W rows n0+r (+16,+32,+48)
  const long ws16 = (long)16 * K;
  v8f acc[2][4] = {};
  for (int k = 0; k < K; k += 32) {
    v8h a0lo = *(const v8h*)ap0;  v8h a0hi = *(const v8h*)(ap0 + 8);
    v8h a1lo = *(const v8h*)ap1;  v8h a1hi = *(const v8h*)(ap1 + 8);
    v16h av0 = SHUF16(a0lo, a0hi);
    v16h av1 = SHUF16(a1lo, a1hi);
    v16h bv0 = *(const v16h*)(bp);
    v16h bv1 = *(const v16h*)(bp + ws16);
    v16h bv2 = *(const v16h*)(bp + 2 * ws16);
    v16h bv3 = *(const v16h*)(bp + 3 * ws16);
    __builtin_prefetch((const void*)(bp + 512), 0, 0);        // stream weights ahead
    acc[0][0] = WMMA(av0, bv0, acc[0][0]);
    acc[0][1] = WMMA(av0, bv1, acc[0][1]);
    acc[0][2] = WMMA(av0, bv2, acc[0][2]);
    acc[0][3] = WMMA(av0, bv3, acc[0][3]);
    acc[1][0] = WMMA(av1, bv0, acc[1][0]);
    acc[1][1] = WMMA(av1, bv1, acc[1][1]);
    acc[1][2] = WMMA(av1, bv2, acc[1][2]);
    acc[1][3] = WMMA(av1, bv3, acc[1][3]);
    ap0 += 32; ap1 += 32; bp += 32;
  }
  #pragma unroll
  for (int mi = 0; mi < 2; mi++) {
    #pragma unroll
    for (int ni = 0; ni < 4; ni++) {
      int col = n0 + ni * 16 + r;
      float bval = bias ? bias[col] : 0.f;
      #pragma unroll
      for (int i = 0; i < 8; i++) {
        long m = (long)m0 + mi * 16 + hi * 8 + i;
        long idx = m * (long)N + col;
        float v = acc[mi][ni][i];
        if (mode == 0) {
          v = apply_act(v + bval, act);
          if (C16) C16[idx] = (half_t)v;
          else     C[idx] = v;
        } else if (mode == 1) {
          C[idx] += v;
        } else {
          C[idx] += gate[m * E_ + gate_e] * (v + bval);
        }
      }
    }
  }
}

// ---------------------------------------------------------------------------
// Small WMMA GEMM (N=16/32): one 16x16 tile per wave.
// ---------------------------------------------------------------------------
__global__ void k_gemm_small(const half_t* __restrict__ A, const half_t* __restrict__ W,
                             const float* __restrict__ bias, float* __restrict__ C,
                             int M, int N, int K, int act) {
  int wave = threadIdx.x >> 5;
  int lane = threadIdx.x & 31;
  int n0 = (blockIdx.x * 4 + wave) * 16;
  int m0 = blockIdx.y * 16;
  if (n0 >= N) return;                       // wave-uniform
  int r  = lane & 15;
  int hi = lane >> 4;
  const half_t* ap = A + (long)(m0 + r) * K + hi * 16;
  const half_t* bp = W + (long)(n0 + r) * K + hi * 16;
  v8f acc = {};
  for (int k = 0; k < K; k += 32) {
    v8h alo = *(const v8h*)ap;
    v8h ahi = *(const v8h*)(ap + 8);
    v16h av = SHUF16(alo, ahi);
    v16h bv = *(const v16h*)bp;
    acc = WMMA(av, bv, acc);
    ap += 32; bp += 32;
  }
  int col = n0 + r;
  float bval = bias ? bias[col] : 0.f;
  #pragma unroll
  for (int i = 0; i < 8; i++) {
    long m = (long)m0 + hi * 8 + i;
    C[m * (long)N + col] = apply_act(acc[i] + bval, act);
  }
}

// ---------------------------------------------------------------------------
// Depthwise causal conv (KC=4) over sequence; reads xi half of xz in place
// ---------------------------------------------------------------------------
__global__ void k_conv(const float* __restrict__ xz, const float* __restrict__ cw,
                       const float* __restrict__ cb, float* __restrict__ xi,
                       half_t* __restrict__ xih) {
  long i = (long)blockIdx.x * blockDim.x + threadIdx.x;   // M_*DI_
  int c = (int)(i & (DI_ - 1));
  long m = i >> 10;
  int s = (int)(m & (S_ - 1));
  long b = m >> 10;
  float acc = cb[c];
  #pragma unroll
  for (int k = 0; k < KC_; k++) {
    int sp = s - (KC_ - 1) + k;
    if (sp >= 0) acc += cw[c * KC_ + k] * xz[((b * S_ + sp) * (long)(2 * DI_)) + c];
  }
  xi[i] = acc;
  xih[i] = (half_t)acc;
}

// ---------------------------------------------------------------------------
// a,b terms: a[m,c]=sum_j exp(delta[m,j]*(-exp(A_log[c,j]))),
//            b[m,c]=xi[m,c]*sum_j delta[m,j]*B_param[m,j]
// ---------------------------------------------------------------------------
__global__ void k_ab(const float* __restrict__ delta, const float* __restrict__ xdbl,
                     const float* __restrict__ Alog, const float* __restrict__ xi,
                     float* __restrict__ abuf, float* __restrict__ bbuf) {
  long i = (long)blockIdx.x * blockDim.x + threadIdx.x;   // M_*DI_
  int c = (int)(i & (DI_ - 1));
  long m = i >> 10;
  const float* dl = delta + m * DS_;
  const float* Bp = xdbl + m * 32 + DS_;
  const float* Ac = Alog + (long)c * DS_;
  float a = 0.f, dot = 0.f;
  #pragma unroll
  for (int j = 0; j < DS_; j++) {
    float dj = dl[j];
    a += expf(dj * (-expf(Ac[j])));
    dot += dj * Bp[j];
  }
  abuf[i] = a;
  bbuf[i] = xi[i] * dot;
}

// ---------------------------------------------------------------------------
// Selective scan: y_s = a_s*y_{s-1} + b_s, per (batch, channel)
// ---------------------------------------------------------------------------
__global__ void k_scan(const float* __restrict__ abuf, const float* __restrict__ bbuf,
                       float* __restrict__ ybuf) {
  int t = blockIdx.x * blockDim.x + threadIdx.x;          // B_*DI_
  if (t >= B_ * DI_) return;
  int c = t & (DI_ - 1);
  int b = t >> 10;
  float y = 0.f;
  for (int s = 0; s < S_; s++) {
    long idx = ((long)(b * S_ + s)) * DI_ + c;
    y = abuf[idx] * y + bbuf[idx];
    ybuf[idx] = y;
  }
}

// ---------------------------------------------------------------------------
// y *= silu(z); emit fp16 for out_proj GEMM
// ---------------------------------------------------------------------------
__global__ void k_siluz(const float* __restrict__ ybuf, const float* __restrict__ xz,
                        half_t* __restrict__ yh) {
  long i = (long)blockIdx.x * blockDim.x + threadIdx.x;   // M_*DI_
  int c = (int)(i & (DI_ - 1));
  long m = i >> 10;
  float z = xz[m * (long)(2 * DI_) + DI_ + c];
  float sil = z / (1.f + expf(-z));
  yh[i] = (half_t)(ybuf[i] * sil);
}

// ---------------------------------------------------------------------------
// Router: logits -> softmax -> top2 -> normalized gate weights + aux stats
// ---------------------------------------------------------------------------
__global__ void k_router(const float* __restrict__ xn, const float* __restrict__ rw,
                         const float* __restrict__ rb, float* __restrict__ gate,
                         float* __restrict__ fsum, float* __restrict__ psum) {
  int m = blockIdx.x * blockDim.x + threadIdx.x;
  if (m >= M_) return;
  const float* xr = xn + (long)m * D_;
  float lg[E_];
  for (int e = 0; e < E_; e++) {
    const float* wr = rw + (long)e * D_;
    float s = rb[e];
    for (int d = 0; d < D_; d++) s += xr[d] * wr[d];
    lg[e] = s;
  }
  float mx = lg[0];
  for (int e = 1; e < E_; e++) mx = fmaxf(mx, lg[e]);
  float p[E_]; float den = 0.f;
  for (int e = 0; e < E_; e++) { p[e] = expf(lg[e] - mx); den += p[e]; }
  for (int e = 0; e < E_; e++) p[e] /= den;
  int i1 = 0;
  for (int e = 1; e < E_; e++) if (p[e] > p[i1]) i1 = e;
  int i2 = (i1 == 0) ? 1 : 0;
  for (int e = 0; e < E_; e++) if (e != i1 && p[e] > p[i2]) i2 = e;
  float tsum = p[i1] + p[i2];
  for (int e = 0; e < E_; e++) gate[(long)m * E_ + e] = 0.f;
  gate[(long)m * E_ + i1] = p[i1] / tsum;
  gate[(long)m * E_ + i2] = p[i2] / tsum;
  for (int e = 0; e < E_; e++) atomicAdd(&psum[e], p[e]);
  atomicAdd(&fsum[i1], 1.f);
  atomicAdd(&fsum[i2], 1.f);
}

__global__ void k_aux(const float* __restrict__ fsum, const float* __restrict__ psum,
                      float* __restrict__ aux) {
  if (threadIdx.x == 0) {
    float s = 0.f;
    for (int e = 0; e < E_; e++)
      s += (fsum[e] / (float)(M_ * 2)) * (psum[e] / (float)M_);
    aux[0] += (float)E_ * s;
  }
}

// ---------------------------------------------------------------------------
// Host orchestration
// ---------------------------------------------------------------------------
extern "C" void kernel_launch(void* const* d_in, const int* in_sizes, int n_in,
                              void* d_out, int out_size, void* d_ws, size_t ws_size,
                              hipStream_t stream) {
  (void)in_sizes; (void)n_in; (void)out_size; (void)ws_size;
  const int*   ids     = (const int*)  d_in[0];
  const float* tok_emb = (const float*)d_in[1];
  const float* pos_emb = (const float*)d_in[2];
  const float* norm1_w = (const float*)d_in[3];
  const float* norm2_w = (const float*)d_in[4];
  const float* in_w    = (const float*)d_in[5];
  const float* conv_w  = (const float*)d_in[6];
  const float* conv_b  = (const float*)d_in[7];
  const float* xproj_w = (const float*)d_in[8];
  const float* dt_w    = (const float*)d_in[9];
  const float* dt_b    = (const float*)d_in[10];
  const float* A_log   = (const float*)d_in[11];
  const float* out_w   = (const float*)d_in[13];
  const float* rout_w  = (const float*)d_in[14];
  const float* rout_b  = (const float*)d_in[15];
  const float* w1      = (const float*)d_in[16];
  const float* b1      = (const float*)d_in[17];
  const float* w2      = (const float*)d_in[18];
  const float* b2      = (const float*)d_in[19];
  const float* fnorm_w = (const float*)d_in[20];
  const float* lm_w    = (const float*)d_in[21];
  float* out = (float*)d_out;

  // workspace carve-up
  char* ws = (char*)d_ws;
  size_t off = 0;
  auto carve = [&](size_t bytes) -> void* {
    void* p = ws + off; off += (bytes + 255) & ~(size_t)255; return p;
  };
  float*  x    = (float*) carve((size_t)M_ * D_ * 4);
  float*  xn   = (float*) carve((size_t)M_ * D_ * 4);
  float*  xz   = (float*) carve((size_t)M_ * 2 * DI_ * 4);
  float*  xi   = (float*) carve((size_t)M_ * DI_ * 4);
  float*  abuf = (float*) carve((size_t)M_ * DI_ * 4);
  float*  bbuf = (float*) carve((size_t)M_ * DI_ * 4);
  float*  ybuf = (float*) carve((size_t)M_ * DI_ * 4);
  float*  xdbl = (float*) carve((size_t)M_ * 32 * 4);
  float*  dlt  = (float*) carve((size_t)M_ * DS_ * 4);
  float*  gate = (float*) carve((size_t)M_ * E_ * 4);
  float*  fsum = (float*) carve(256);
  float*  psum = (float*) carve(256);
  half_t* xn16 = (half_t*)carve((size_t)M_ * D_ * 2);
  half_t* a16  = (half_t*)carve((size_t)M_ * DFF_ * 2);   // xi16 / y16 / h16
  half_t* w16  = (half_t*)carve((size_t)V_ * D_ * 2);     // weight staging (max: lm_head)

  auto cvt = [&](const float* src, half_t* dst, long n) {
    k_cvt16<<<dim3((unsigned)((n + 255) / 256)), 256, 0, stream>>>(src, dst, n);
  };
  // big GEMM: M%32==0, N%256==0, K%32==0
  auto gemm = [&](const half_t* A, const half_t* W, const float* bias, float* C,
                  half_t* C16, const float* g, int ge, int M, int N, int K,
                  int mode, int act) {
    dim3 grid(N / 256, M / 32);
    k_gemm_big<<<grid, 128, 0, stream>>>(A, W, bias, C, C16, g, ge, M, N, K, mode, act);
  };
  auto gemm_sm = [&](const half_t* A, const half_t* W, const float* bias, float* C,
                     int M, int N, int K, int act) {
    dim3 grid((N + 63) / 64, M / 16);
    k_gemm_small<<<grid, 128, 0, stream>>>(A, W, bias, C, M, N, K, act);
  };

  // init aux accumulator + embeddings
  k_zero<<<1, 1, 0, stream>>>(out + AUX_IDX, 1);
  k_embed<<<(M_ * D_) / 256, 256, 0, stream>>>(ids, tok_emb, pos_emb, x);

  for (int l = 0; l < NL_; l++) {
    // ---- Mamba block ----
    k_rmsnorm<<<M_, 256, 0, stream>>>(x, norm1_w + (long)l * D_, nullptr, xn16);

    cvt(in_w + (long)l * 2 * DI_ * D_, w16, (long)2 * DI_ * D_);
    gemm(xn16, w16, nullptr, xz, nullptr, nullptr, 0, M_, 2 * DI_, D_, 0, 0);  // in_proj

    k_conv<<<(M_ * DI_) / 256, 256, 0, stream>>>(
        xz, conv_w + (long)l * DI_ * KC_, conv_b + (long)l * DI_, xi, a16);

    cvt(xproj_w + (long)l * 2 * DS_ * DI_, w16, (long)2 * DS_ * DI_);
    gemm_sm(a16, w16, nullptr, xdbl, M_, 2 * DS_, DI_, 0);                     // x_proj

    cvt(dt_w + (long)l * DS_ * DI_, w16, (long)DS_ * DI_);
    gemm_sm(a16, w16, dt_b + (long)l * DS_, dlt, M_, DS_, DI_, 2);             // dt+softplus

    k_ab<<<(M_ * DI_) / 256, 256, 0, stream>>>(dlt, xdbl, A_log + (long)l * DI_ * DS_,
                                               xi, abuf, bbuf);
    k_scan<<<(B_ * DI_) / 256, 256, 0, stream>>>(abuf, bbuf, ybuf);
    k_siluz<<<(M_ * DI_) / 256, 256, 0, stream>>>(ybuf, xz, a16);

    cvt(out_w + (long)l * D_ * DI_, w16, (long)D_ * DI_);
    gemm(a16, w16, nullptr, x, nullptr, nullptr, 0, M_, D_, DI_, 1, 0);        // out_proj += res

    // ---- MoE block ----
    k_rmsnorm<<<M_, 256, 0, stream>>>(x, norm2_w + (long)l * D_, xn, xn16);
    k_zero<<<1, 32, 0, stream>>>(fsum, E_);
    k_zero<<<1, 32, 0, stream>>>(psum, E_);
    k_router<<<M_ / 256, 256, 0, stream>>>(xn, rout_w + (long)l * E_ * D_,
                                           rout_b + (long)l * E_, gate, fsum, psum);
    for (int e = 0; e < E_; e++) {
      long we = (long)(l * E_ + e);
      cvt(w1 + we * DFF_ * D_, w16, (long)DFF_ * D_);
      gemm(xn16, w16, b1 + we * DFF_, nullptr, a16, nullptr, 0,
           M_, DFF_, D_, 0, 1);                                  // h = gelu(.) -> fp16 direct
      cvt(w2 + we * D_ * DFF_, w16, (long)D_ * DFF_);
      gemm(a16, w16, b2 + we * D_, x, nullptr, gate, e,
           M_, D_, DFF_, 2, 0);                                  // x += g*(h*w2+b2)
    }
    k_aux<<<1, 32, 0, stream>>>(fsum, psum, out + AUX_IDX);
  }

  // ---- final norm + lm_head ----
  k_rmsnorm<<<M_, 256, 0, stream>>>(x, fnorm_w, nullptr, xn16);
  cvt(lm_w, w16, (long)V_ * D_);
  gemm(xn16, w16, nullptr, out, nullptr, nullptr, 0, M_, V_, D_, 0, 0);
}
